// ThalamusGateway_80882824118915
// MI455X (gfx1250) — matrix operand order repaired and checked by reading, not compiled, not run on hardware
//
#include <hip/hip_runtime.h>
#include <hip/hip_bf16.h>
#include <math.h>

// ---------------------------------------------------------------------------
// Problem constants (from reference setup_inputs)
// ---------------------------------------------------------------------------
#define BB   4
#define SS   8192
#define DD   1024
#define HH   512
#define CC   64
#define KK   409          // top-k
#define GATE 0.3f         // alpha_mod(=1.0 at phase 0) * OSC_COUPLING

#define ROWS   (BB * SS)          // 32768
#define NT1    (HH / 16)          // 32 column tiles of W1
#define KT     (DD / 32)          // 32 K-steps of 32
#define NTR    (CC / 16)          // 4 column tiles of Wr
#define TPB    26                 // ceil(409/16) row tiles per batch

// Workspace layout (bytes)
#define WS_SCORES   0u                          // ROWS floats      (128 KB)
#define WS_W1BF     131072u                     // DD*HH bf16       (1 MB)
#define WS_WRBF     1179648u                    // DD*CC bf16       (128 KB)
#define WS_IDX      1310720u                    // BB*KK int        (6.4 KB)

typedef __attribute__((ext_vector_type(16))) __bf16          v16bf;
typedef __attribute__((ext_vector_type(8)))  float           v8f;

__device__ __forceinline__ v8f wmma_bf16(v16bf a, v16bf b, v8f c) {
  return __builtin_amdgcn_wmma_f32_16x16x32_bf16(
      false, a, false, b, (short)0, c, false, false);
}

// Build the A operand (16x32 bf16 tile) for this lane per ISA 7.12.2:
// lane<16 : row=lane,    elems 0..7 = K[base..base+7],  8..15 = K[base+16..23]
// lane>=16: row=lane-16, elems 0..7 = K[base+8..+15],   8..15 = K[base+24..31]
// Uses native fp32->bf16 converts (v_cvt_pk_bf16_f32) instead of integer packs.
__device__ __forceinline__ v16bf load_a_tile(const float* rowPtr, int kbase,
                                             int khalf, float4* raw /*out 4*/) {
  const float4* p = (const float4*)(rowPtr + kbase + khalf);
  float4 a0 = p[0], a1 = p[1];
  const float4* q = (const float4*)(rowPtr + kbase + 16 + khalf);
  float4 b0 = q[0], b1 = q[1];
  raw[0] = a0; raw[1] = a1; raw[2] = b0; raw[3] = b1;
  v16bf v;
  v[0]  = (__bf16)a0.x; v[1]  = (__bf16)a0.y; v[2]  = (__bf16)a0.z; v[3]  = (__bf16)a0.w;
  v[4]  = (__bf16)a1.x; v[5]  = (__bf16)a1.y; v[6]  = (__bf16)a1.z; v[7]  = (__bf16)a1.w;
  v[8]  = (__bf16)b0.x; v[9]  = (__bf16)b0.y; v[10] = (__bf16)b0.z; v[11] = (__bf16)b0.w;
  v[12] = (__bf16)b1.x; v[13] = (__bf16)b1.y; v[14] = (__bf16)b1.z; v[15] = (__bf16)b1.w;
  return v;
}

// ---------------------------------------------------------------------------
// Kernel 1: swizzle a row-major fp32 [DD x N] weight into bf16 WMMA-B layout:
//   out[((kt*NT + nt)*32 + lane)*16 + e] = W[k][n]
//   lane<16: n = nt*16+lane,    k = kt*32 + (e&7) + (e>=8 ? 16 : 0)
//   lane>=16:n = nt*16+lane-16, k = kt*32 + 8 + (e&7) + (e>=8 ? 16 : 0)
// One 32B contiguous load per lane per WMMA in the GEMM kernels.
// ---------------------------------------------------------------------------
__global__ void prep_b_kernel(const float* __restrict__ W,
                              __bf16* __restrict__ out,
                              int N, int NT, int total) {
  int tid = blockIdx.x * blockDim.x + threadIdx.x;
  if (tid >= total) return;
  int e    = tid & 15;
  int lane = (tid >> 4) & 31;
  int rest = tid >> 9;
  int nt   = rest % NT;
  int kt   = rest / NT;
  int k = kt * 32 + ((lane >= 16) ? 8 : 0) + (e & 7) + ((e >= 8) ? 16 : 0);
  int n = nt * 16 + (lane & 15);
  out[tid] = (__bf16)W[(size_t)k * N + n];
}

// ---------------------------------------------------------------------------
// Kernel 2: fused  scores = GELU(x@W1 + b1) @ W2 + b2 - GATE
// 256 threads = 8 waves. Block = 16 rows. Wave w owns H-cols [64w, 64w+64).
// ---------------------------------------------------------------------------
__global__ void __launch_bounds__(256)
scores_kernel(const float* __restrict__ x,
              const __bf16* __restrict__ w1bf,
              const float* __restrict__ b1,
              const float* __restrict__ W2,
              const float* __restrict__ b2,
              float* __restrict__ scores) {
  __shared__ float bsum[16];
  const int tid  = threadIdx.x;
  const int lane = tid & 31;
  const int w    = tid >> 5;
  const int rb   = blockIdx.x * 16;

  if (tid < 16) bsum[tid] = 0.0f;
  __syncthreads();

  const int   m16   = lane & 15;
  const int   khalf = (lane >= 16) ? 8 : 0;
  const float* rowPtr = x + (size_t)(rb + m16) * DD;
  const v16bf* Bp = (const v16bf*)w1bf;

  v8f acc0 = {0,0,0,0,0,0,0,0};
  v8f acc1 = acc0, acc2 = acc0, acc3 = acc0;

  float4 raw[4];
#pragma unroll 2
  for (int kt = 0; kt < KT; ++kt) {
    v16bf a = load_a_tile(rowPtr, kt * 32, khalf, raw);
    int base = (kt * NT1 + w * 4) * 32 + lane;
    v16bf bv0 = Bp[base +  0];
    v16bf bv1 = Bp[base + 32];
    v16bf bv2 = Bp[base + 64];
    v16bf bv3 = Bp[base + 96];
    acc0 = wmma_bf16(a, bv0, acc0);
    acc1 = wmma_bf16(a, bv1, acc1);
    acc2 = wmma_bf16(a, bv2, acc2);
    acc3 = wmma_bf16(a, bv3, acc3);
  }

  // Epilogue: bias + exact GELU + dot with W2 column weights.
  float s[8];
#pragma unroll
  for (int j = 0; j < 8; ++j) s[j] = 0.0f;
  v8f* accs[4] = {&acc0, &acc1, &acc2, &acc3};
#pragma unroll
  for (int t = 0; t < 4; ++t) {
    int col = w * 64 + t * 16 + m16;
    float b1v = b1[col];
    float w2v = W2[col];
#pragma unroll
    for (int j = 0; j < 8; ++j) {
      float h = (*accs[t])[j] + b1v;
      float g = 0.5f * h * (1.0f + erff(h * 0.70710678118654752f));
      s[j] += g * w2v;
    }
  }
  // Butterfly reduce across the 16 lanes (columns) of each half-wave.
#pragma unroll
  for (int off = 1; off < 16; off <<= 1)
#pragma unroll
    for (int j = 0; j < 8; ++j) s[j] += __shfl_xor(s[j], off, 32);

  if (m16 == 0) {
    int mbase = (lane >= 16) ? 8 : 0;
#pragma unroll
    for (int j = 0; j < 8; ++j) atomicAdd(&bsum[mbase + j], s[j]);
  }
  __syncthreads();
  if (tid < 16) scores[rb + tid] = bsum[tid] + b2[0] - GATE;
}

// ---------------------------------------------------------------------------
// Kernel 3: exact top-k via rank counting; emits indices in descending-score,
// index-ascending order (matches jax.lax.top_k).
// ---------------------------------------------------------------------------
__global__ void __launch_bounds__(256)
topk_kernel(const float* __restrict__ scores, int* __restrict__ idxOut) {
  __shared__ float sc[SS];
  const int b     = blockIdx.x / (SS / 256);
  const int chunk = blockIdx.x % (SS / 256);
  const int tid   = threadIdx.x;
  for (int j = tid; j < SS; j += 256) sc[j] = scores[b * SS + j];
  __syncthreads();
  const int   i  = chunk * 256 + tid;
  const float si = sc[i];
  int cnt = 0;
  for (int j = 0; j < SS; ++j) {
    float v = sc[j];
    cnt += (v > si) || (v == si && j < i);
  }
  if (cnt < KK) idxOut[b * KK + cnt] = i;
}

// ---------------------------------------------------------------------------
// Kernel 4: gather filtered rows (written to output) + router GEMM via WMMA +
// softmax * syn. One wave per 16-row tile; 26 tiles per batch.
// ---------------------------------------------------------------------------
__global__ void __launch_bounds__(32)
router_kernel(const float* __restrict__ x,
              const __bf16* __restrict__ wrbf,
              const float* __restrict__ br,
              const float* __restrict__ syn,
              const int* __restrict__ idx,
              float* __restrict__ outFiltered,
              float* __restrict__ outWeighted) {
  const int b    = blockIdx.x / TPB;
  const int tile = blockIdx.x % TPB;
  const int lane = threadIdx.x;
  const int m16  = lane & 15;
  const int khalf = (lane >= 16) ? 8 : 0;

  const int grow  = tile * 16 + m16;           // gather-row for A load/store
  const bool validA = (grow < KK);
  const int gi = idx[b * KK + (validA ? grow : (KK - 1))];

  const float* rowPtr = x + ((size_t)b * SS + gi) * DD;
  float* fOut = outFiltered + ((size_t)b * KK + (validA ? grow : 0)) * DD;
  const v16bf* Bp = (const v16bf*)wrbf;

  v8f acc0 = {0,0,0,0,0,0,0,0};
  v8f acc1 = acc0, acc2 = acc0, acc3 = acc0;

  float4 raw[4];
#pragma unroll 2
  for (int kt = 0; kt < KT; ++kt) {
    v16bf a = load_a_tile(rowPtr, kt * 32, khalf, raw);
    if (validA) {  // stream gathered fp32 row straight to `filtered`
      float4* d0 = (float4*)(fOut + kt * 32 + khalf);
      d0[0] = raw[0]; d0[1] = raw[1];
      float4* d1 = (float4*)(fOut + kt * 32 + 16 + khalf);
      d1[0] = raw[2]; d1[1] = raw[3];
    }
    int base = kt * NTR * 32 + lane;
    v16bf bv0 = Bp[base +  0];
    v16bf bv1 = Bp[base + 32];
    v16bf bv2 = Bp[base + 64];
    v16bf bv3 = Bp[base + 96];
    acc0 = wmma_bf16(a, bv0, acc0);
    acc1 = wmma_bf16(a, bv1, acc1);
    acc2 = wmma_bf16(a, bv2, acc2);
    acc3 = wmma_bf16(a, bv3, acc3);
  }

  // logits -> softmax over the 64 router columns, per row.
  v8f* accs[4] = {&acc0, &acc1, &acc2, &acc3};
  float l[4][8];
#pragma unroll
  for (int t = 0; t < 4; ++t) {
    float brv = br[t * 16 + m16];
#pragma unroll
    for (int j = 0; j < 8; ++j) l[t][j] = (*accs[t])[j] + brv;
  }
  float rm[8];
#pragma unroll
  for (int j = 0; j < 8; ++j) {
    rm[j] = fmaxf(fmaxf(l[0][j], l[1][j]), fmaxf(l[2][j], l[3][j]));
#pragma unroll
    for (int off = 1; off < 16; off <<= 1)
      rm[j] = fmaxf(rm[j], __shfl_xor(rm[j], off, 32));
  }
  float es[8];
#pragma unroll
  for (int j = 0; j < 8; ++j) {
    float acc = 0.0f;
#pragma unroll
    for (int t = 0; t < 4; ++t) {
      l[t][j] = __expf(l[t][j] - rm[j]);
      acc += l[t][j];
    }
#pragma unroll
    for (int off = 1; off < 16; off <<= 1) acc += __shfl_xor(acc, off, 32);
    es[j] = acc;
  }
  const int mbase = (lane >= 16) ? 8 : 0;
#pragma unroll
  for (int t = 0; t < 4; ++t) {
    int col = t * 16 + m16;
    float sy = syn[col];
#pragma unroll
    for (int j = 0; j < 8; ++j) {
      int rowC = tile * 16 + mbase + j;     // C-layout row for this lane/VGPR
      if (rowC < KK)
        outWeighted[((size_t)b * KK + rowC) * CC + col] = (l[t][j] / es[j]) * sy;
    }
  }
}

// ---------------------------------------------------------------------------
extern "C" void kernel_launch(void* const* d_in, const int* in_sizes, int n_in,
                              void* d_out, int out_size, void* d_ws, size_t ws_size,
                              hipStream_t stream) {
  const float* x   = (const float*)d_in[0];
  const float* W1  = (const float*)d_in[1];
  const float* b1  = (const float*)d_in[2];
  const float* W2  = (const float*)d_in[3];
  const float* b2  = (const float*)d_in[4];
  const float* Wr  = (const float*)d_in[5];
  const float* br  = (const float*)d_in[6];
  const float* syn = (const float*)d_in[7];
  (void)in_sizes; (void)n_in; (void)out_size; (void)ws_size;

  char* ws = (char*)d_ws;
  float*   scores = (float*)(ws + WS_SCORES);
  __bf16*  w1bf   = (__bf16*)(ws + WS_W1BF);
  __bf16*  wrbf   = (__bf16*)(ws + WS_WRBF);
  int*     idxBuf = (int*)(ws + WS_IDX);

  float* outFiltered = (float*)d_out;
  float* outWeighted = outFiltered + (size_t)BB * KK * DD;

  // 1) weight swizzle to bf16 WMMA-B layout
  prep_b_kernel<<<(DD * HH + 255) / 256, 256, 0, stream>>>(W1, w1bf, HH, NT1, DD * HH);
  prep_b_kernel<<<(DD * CC + 255) / 256, 256, 0, stream>>>(Wr, wrbf, CC, NTR, DD * CC);

  // 2) fused relevance-MLP scores (big bf16 WMMA GEMM)
  scores_kernel<<<ROWS / 16, 256, 0, stream>>>(x, w1bf, b1, W2, b2, scores);

  // 3) top-k indices, descending order
  topk_kernel<<<BB * (SS / 256), 256, 0, stream>>>(scores, idxBuf);

  // 4) gather + router WMMA GEMM + softmax*syn
  router_kernel<<<BB * TPB, 32, 0, stream>>>(x, wrbf, br, syn, idxBuf,
                                             outFiltered, outWeighted);
}
